// Net_27127013441773
// MI455X (gfx1250) — 8-bit matrix operand order repaired and checked by reading, compiled, not run on hardware
//
#include <hip/hip_runtime.h>
#include <hip/hip_bf16.h>
#include <cstdint>
#include <cstddef>


typedef __attribute__((ext_vector_type(8))) int v8i;

__global__ __launch_bounds__(128) void lda_kernel(const float* __restrict__ in,
                                                  const float* __restrict__ w,
                                                  const float* __restrict__ bias,
                                                  float* __restrict__ out) {
    __shared__ float spl[120];
    const int blk = blockIdx.x;
    const int b = blk / 298;
    const int j = blk % 298;
    const float* src = in + ((size_t)b * 300 + j) * 40;
    for (int i = threadIdx.x; i < 120; i += blockDim.x) spl[i] = src[i];
    __syncthreads();
    const int o = threadIdx.x;
    if (o < 120) {
        float acc = bias[o];
        const float* wr = w + (size_t)o * 120;
        #pragma unroll 4
        for (int k = 0; k < 120; ++k) acc = fmaf(spl[k], wr[k], acc);
        out[((size_t)b * 298 + j) * 120 + o] = acc;
    }
}

__device__ __forceinline__ void block_absmax_commit(float m, float* slot) {
    __shared__ float red[256];
    red[threadIdx.x] = m;
    __syncthreads();
    for (int s = 128; s > 0; s >>= 1) {
        if ((int)threadIdx.x < s)
            red[threadIdx.x] = fmaxf(red[threadIdx.x], red[threadIdx.x + s]);
        __syncthreads();
    }
    if (threadIdx.x == 0)
        atomicMax((unsigned int*)slot, __float_as_uint(red[0]));
}

__global__ __launch_bounds__(256) void absmax_flat(const float* __restrict__ p,
                                                   long long n, float* slot) {
    float m = 0.f;
    for (long long i = (long long)blockIdx.x * blockDim.x + threadIdx.x; i < n;
         i += (long long)gridDim.x * blockDim.x)
        m = fmaxf(m, fabsf(p[i]));
    block_absmax_commit(m, slot);
}

__global__ __launch_bounds__(256) void absmax_spliced(
    const float* __restrict__ x, int Tin, int F, int d0, int d1, int d2,
    int ss, int Tout, int M, int Kin, float* slot) {
    const size_t total = (size_t)M * (size_t)Kin;
    float m = 0.f;
    for (size_t i = (size_t)blockIdx.x * blockDim.x + threadIdx.x; i < total;
         i += (size_t)gridDim.x * blockDim.x) {
        int mm = (int)(i / (size_t)Kin);
        int k  = (int)(i - (size_t)mm * Kin);
        int c  = k / F;
        int f  = k - c * F;
        int off = (c == 0) ? d0 : ((c == 1) ? d1 : d2);
        int b  = mm / Tout;
        int j  = mm - b * Tout;
        int t  = j * ss + off;
        m = fmaxf(m, fabsf(x[((size_t)b * Tin + t) * F + f]));
    }
    block_absmax_commit(m, slot);
}

__global__ __launch_bounds__(256) void quant_act(
    const float* __restrict__ x, int Tin, int F, int d0, int d1, int d2,
    int ss, int Tout, int M, int Kin, int Kpad,
    const float* __restrict__ slot, signed char* __restrict__ Aq) {
    const size_t total = (size_t)M * (size_t)Kpad;
    const size_t i = (size_t)blockIdx.x * blockDim.x + threadIdx.x;
    if (i >= total) return;
    const int mm = (int)(i / (size_t)Kpad);
    const int k  = (int)(i - (size_t)mm * Kpad);
    signed char q = 0;
    if (k < Kin) {
        const float sx = 127.f / fmaxf(slot[0], 1e-8f);
        int c  = k / F;
        int f  = k - c * F;
        int off = (c == 0) ? d0 : ((c == 1) ? d1 : d2);
        int b  = mm / Tout;
        int j  = mm - b * Tout;
        int t  = j * ss + off;
        float r = rintf(x[((size_t)b * Tin + t) * F + f] * sx);
        r = fminf(fmaxf(r, -127.f), 127.f);
        q = (signed char)(int)r;
    }
    Aq[i] = q;
}

__global__ __launch_bounds__(256) void quant_w(
    const float* __restrict__ w, const float* __restrict__ slot,
    signed char* __restrict__ Wq, int N, int Kin, int Kpad, int Npad) {
    const size_t total = (size_t)Npad * (size_t)Kpad;
    const size_t i = (size_t)blockIdx.x * blockDim.x + threadIdx.x;
    if (i >= total) return;
    const int n = (int)(i / (size_t)Kpad);
    const int k = (int)(i - (size_t)n * Kpad);
    signed char q = 0;
    if (n < N && k < Kin) {
        const float sw = 127.f / fmaxf(slot[0], 1e-8f);
        float r = rintf(w[(size_t)n * Kin + k] * sw);
        r = fminf(fmaxf(r, -127.f), 127.f);
        q = (signed char)(int)r;
    }
    Wq[i] = q;
}

__device__ __forceinline__ v8i load_a_frag(const signed char* arow, int k0) {
    int2 a0 = *(const int2*)(arow + k0);
    int2 a1 = *(const int2*)(arow + k0 + 8);
    int2 a2 = *(const int2*)(arow + k0 + 32);
    int2 a3 = *(const int2*)(arow + k0 + 40);
    v8i v;
    v[0] = a0.x; v[1] = a0.y; v[2] = a1.x; v[3] = a1.y;
    v[4] = a2.x; v[5] = a2.y; v[6] = a3.x; v[7] = a3.y;
    return v;
}
__device__ __forceinline__ v8i load_b_frag(const signed char* brow, int k0) {
    int4 b0 = *(const int4*)(brow + k0);
    int4 b1 = *(const int4*)(brow + k0 + 32);
    v8i v;
    v[0] = b0.x; v[1] = b0.y; v[2] = b0.z; v[3] = b0.w;
    v[4] = b1.x; v[5] = b1.y; v[6] = b1.z; v[7] = b1.w;
    return v;
}

__global__ __launch_bounds__(256) void gemm_iu8(
    const signed char* __restrict__ Aq, const signed char* __restrict__ Wq,
    const float* __restrict__ bias, const float* __restrict__ sc,
    const float* __restrict__ ofs, const float* __restrict__ slotx,
    const float* __restrict__ slotw, float* __restrict__ out,
    int M, int N, int Npad, int Kpad, int do_act) {
    const int wave = threadIdx.x >> 5;
    const int lane = threadIdx.x & 31;
    const int tn2cnt = Npad >> 5;
    const int total_tiles = (M >> 5) * tn2cnt;
    const int tile = blockIdx.x * 8 + wave;
    if (tile >= total_tiles) return;
    const int tm2 = tile / tn2cnt;
    const int tn2 = tile - tm2 * tn2cnt;
    const int m_base = tm2 << 5;
    const int n_base = tn2 << 5;
    const int lr = lane & 15;
    const int lh = (lane >> 4) & 1;

    const signed char* arow0 = Aq + ((size_t)(m_base + lr)) * Kpad + lh * 16;
    const signed char* arow1 = arow0 + (size_t)16 * Kpad;
    const signed char* brow0 = Wq + ((size_t)(n_base + lr)) * Kpad + lh * 16;
    const signed char* brow1 = brow0 + (size_t)16 * Kpad;

    v8i acc00 = {0,0,0,0,0,0,0,0};
    v8i acc01 = {0,0,0,0,0,0,0,0};
    v8i acc10 = {0,0,0,0,0,0,0,0};
    v8i acc11 = {0,0,0,0,0,0,0,0};
    for (int k0 = 0; k0 < Kpad; k0 += 64) {
        const v8i a0 = load_a_frag(arow0, k0);
        const v8i a1 = load_a_frag(arow1, k0);
        const v8i b0 = load_b_frag(brow0, k0);
        const v8i b1 = load_b_frag(brow1, k0);
        acc00 = __builtin_amdgcn_wmma_i32_16x16x64_iu8(true, a0, true, b0, acc00,
                                                       false, false);
        acc01 = __builtin_amdgcn_wmma_i32_16x16x64_iu8(true, a0, true, b1, acc01,
                                                       false, false);
        acc10 = __builtin_amdgcn_wmma_i32_16x16x64_iu8(true, a1, true, b0, acc10,
                                                       false, false);
        acc11 = __builtin_amdgcn_wmma_i32_16x16x64_iu8(true, a1, true, b1, acc11,
                                                       false, false);
    }

    const float amx = fmaxf(slotx[0], 1e-8f);
    const float amw = fmaxf(slotw[0], 1e-8f);
    const float inv = (amx * amw) * (1.0f / (127.f * 127.f));
    #pragma unroll
    for (int sn = 0; sn < 2; ++sn) {
        const int n = n_base + sn * 16 + lr;
        if (n < N) {
            const float bn = bias[n];
            const float s = do_act ? sc[n] : 1.f;
            const float o = do_act ? ofs[n] : 0.f;
            #pragma unroll
            for (int sm = 0; sm < 2; ++sm) {
                const v8i acc = (sn == 0) ? (sm == 0 ? acc00 : acc10)
                                          : (sm == 0 ? acc01 : acc11);
                const int m0 = m_base + sm * 16 + lh * 8;
                #pragma unroll
                for (int r = 0; r < 8; ++r) {
                    float v = (float)acc[r] * inv + bn;
                    if (do_act) v = fmaxf(v, 0.f) * s + o;
                    out[(size_t)(m0 + r) * N + n] = v;
                }
            }
        }
    }
}

extern "C" void kernel_launch(void* const* d_in, const int* in_sizes, int n_in,
                              void* d_out, int out_size, void* d_ws, size_t ws_size,
                              hipStream_t stream) {
    (void)in_sizes; (void)n_in; (void)out_size; (void)ws_size;
    const float* input = (const float*)d_in[0];
    const float* lda_w = (const float*)d_in[1];
    const float* lda_b = (const float*)d_in[2];
    float* out = (float*)d_out;

    char* ws = (char*)d_ws;
    float* slots = (float*)ws;
    size_t off = 256;
    float* actA = (float*)(ws + off); off += (size_t)32 * 1024 * 1024;
    float* actB = (float*)(ws + off); off += (size_t)32 * 1024 * 1024;
    signed char* Aq = (signed char*)(ws + off); off += (size_t)25 * 1024 * 1024;
    signed char* Wq = (signed char*)(ws + off); off += (size_t)4 * 1024 * 1024;

    hipMemsetAsync(slots, 0, 256, stream);

    lda_kernel<<<128 * 298, 128, 0, stream>>>(input, lda_w, lda_b, actA);

    const int Tin[9]  = {298, 298, 99, 97, 91, 85, 79, 73, 73};
    const int Toutv[9]= {298,  99, 97, 91, 85, 79, 73, 73, 73};
    const int Cn[9]   = {1, 3, 3, 3, 3, 3, 3, 1, 1};
    const int dstep[9]= {0, 1, 1, 3, 3, 3, 3, 0, 0};
    const int ssv[9]  = {1, 3, 1, 1, 1, 1, 1, 1, 1};
    const int Fv[9]   = {120, 625, 625, 625, 625, 625, 625, 625, 625};
    const int Nv[9]   = {625, 625, 625, 625, 625, 625, 625, 625, 4664};
    const float* srcs[9] = {actA, out, actA, actB, actA, actB, actA, actB, actA};
    float*       dsts[9] = {(float*)out, actA, actB, actA, actB, actA, actB, actA, out};

    for (int i = 0; i < 9; ++i) {
        const float* w  = (const float*)d_in[3 + 2 * i];
        const float* b  = (const float*)d_in[4 + 2 * i];
        const float* sc = (i < 8) ? (const float*)d_in[21 + 2 * i] : nullptr;
        const float* oo = (i < 8) ? (const float*)d_in[22 + 2 * i] : nullptr;
        const int Kin  = Cn[i] * Fv[i];
        const int Kpad = (Kin + 127) & ~127;
        const int M    = 128 * Toutv[i];
        const int N    = Nv[i];
        const int Npad = (N + 31) & ~31;
        float* sx = slots + 2 * i;
        float* sw = slots + 2 * i + 1;

        {
            size_t tot = (size_t)M * (size_t)Kin;
            int blocks = (int)((tot + 255) / 256);
            if (blocks > 8192) blocks = 8192;
            absmax_spliced<<<blocks, 256, 0, stream>>>(
                srcs[i], Tin[i], Fv[i], 0, dstep[i], 2 * dstep[i], ssv[i],
                Toutv[i], M, Kin, sx);
        }
        {
            long long tot = (long long)N * Kin;
            int blocks = (int)((tot + 255) / 256);
            if (blocks > 8192) blocks = 8192;
            absmax_flat<<<blocks, 256, 0, stream>>>(w, tot, sw);
        }
        {
            size_t tot = (size_t)M * (size_t)Kpad;
            quant_act<<<(int)((tot + 255) / 256), 256, 0, stream>>>(
                srcs[i], Tin[i], Fv[i], 0, dstep[i], 2 * dstep[i], ssv[i],
                Toutv[i], M, Kin, Kpad, sx, Aq);
        }
        {
            size_t tot = (size_t)Npad * (size_t)Kpad;
            quant_w<<<(int)((tot + 255) / 256), 256, 0, stream>>>(
                w, sw, Wq, N, Kin, Kpad, Npad);
        }
        {
            int tiles = (M >> 5) * (Npad >> 5);
            gemm_iu8<<<(tiles + 7) / 8, 256, 0, stream>>>(
                Aq, Wq, b, sc, oo, sx, sw, dsts[i], M, N, Npad, Kpad,
                (i < 8) ? 1 : 0);
        }
    }
}
